// RingTransformer_90615220011352
// MI455X (gfx1250) — matrix operand order repaired and checked by reading, not compiled, not run on hardware
//
#include <hip/hip_runtime.h>
#include <math.h>

// ---------------------------------------------------------------------------
// MI455X (gfx1250) transformer forward.  ~410 GFLOP vs ~0.5 GB unique bytes
// -> strongly compute-bound, so every GEMM runs on V_WMMA_F32_16X16X32_BF16
// (bf16 inputs, fp32 accumulate).  Weights are converted + transposed once so
// both GEMM tiles stage as contiguous b128 copies; tiles move to LDS via
// GLOBAL_LOAD_ASYNC_TO_LDS_B128 (ASYNCcnt) with double buffering.
// ---------------------------------------------------------------------------

typedef __attribute__((ext_vector_type(16))) __bf16 v16bf;
typedef __attribute__((ext_vector_type(8)))  float  v8f;

union Frag { v16bf v; uint4 u[2]; };

__device__ __forceinline__ v8f vzero8() {
  v8f v = {0.f,0.f,0.f,0.f,0.f,0.f,0.f,0.f};
  return v;
}

__device__ __forceinline__ unsigned short f32_to_bf16(float f) {
  union { float f; unsigned u; } x; x.f = f;
  unsigned r = x.u + 0x7FFFu + ((x.u >> 16) & 1u);   // round-to-nearest-even
  return (unsigned short)(r >> 16);
}

__device__ __forceinline__ v8f wmma_bf16(v16bf a, v16bf b, v8f c) {
  // D = A(16x32 bf16) * B(32x16 bf16) + C(16x16 f32)
  return __builtin_amdgcn_wmma_f32_16x16x32_bf16(
      false, a, false, b, (short)0, c, false, false);
}

// --- CDNA5 async memory->LDS path (probe via __has_builtin) ----------------
#if defined(__has_builtin)
#if __has_builtin(__builtin_amdgcn_global_load_async_to_lds_b128)
#define USE_ASYNC_LDS 1
#endif
#endif

#ifdef USE_ASYNC_LDS
// Exact parameter types per hipcc diagnostic: pointer to int vector_size(16)
// in global (AS1) / LDS (AS3) address spaces.
typedef int v4i_gcc __attribute__((vector_size(16)));
typedef __attribute__((address_space(1))) v4i_gcc* as1_v4i_ptr;
typedef __attribute__((address_space(3))) v4i_gcc* as3_v4i_ptr;

__device__ __forceinline__ void async_copy16(const void* g, void* l) {
  // GLOBAL_LOAD_ASYNC_TO_LDS_B128: 16B memory -> LDS, tracked by ASYNCcnt
  __builtin_amdgcn_global_load_async_to_lds_b128(
      (as1_v4i_ptr)g, (as3_v4i_ptr)l, 0, 0);
}
__device__ __forceinline__ void wait_async0() {
#if __has_builtin(__builtin_amdgcn_s_wait_asynccnt)
  __builtin_amdgcn_s_wait_asynccnt(0);
#else
  asm volatile("s_wait_asynccnt 0x0" ::: "memory");
#endif
}
#else
__device__ __forceinline__ void wait_async0() {}
#endif

// Model constants
#define NSEQ   2048
#define DIM_   1024
#define DEPTH_ 4
#define HEADS_ 16
#define DH_    64
#define FF_    4096
#define VOCAB_ 32000

// ---------------------------------------------------------------------------
// weight convert fp32[K][N] -> bf16[N][K] (transposed once so GEMM B-tiles
// are contiguous along K; kills scattered ds stores in the GEMM hot loop)
// ---------------------------------------------------------------------------
__global__ __launch_bounds__(256) void cvt_transpose_kernel(
    const float* __restrict__ in, unsigned short* __restrict__ out,
    int K, int N) {
  __shared__ float t[32][33];
  int n0 = blockIdx.x * 32, k0 = blockIdx.y * 32;
  int tx = threadIdx.x, ty = threadIdx.y;  // (32, 8)
#pragma unroll
  for (int j = 0; j < 4; ++j)
    t[ty + 8 * j][tx] = in[(size_t)(k0 + ty + 8 * j) * N + n0 + tx];
  __syncthreads();
#pragma unroll
  for (int j = 0; j < 4; ++j)
    out[(size_t)(n0 + ty + 8 * j) * K + k0 + tx] = f32_to_bf16(t[tx][ty + 8 * j]);
}

// ---------------------------------------------------------------------------
// embedding gather
// ---------------------------------------------------------------------------
__global__ void embed_kernel(const int* __restrict__ tok,
                             const float* __restrict__ emb,
                             float* __restrict__ x) {
  int n = blockIdx.x;
  int t = tok[n];
  const float* src = emb + (size_t)t * DIM_;
  float* dst = x + (size_t)n * DIM_;
  for (int i = threadIdx.x; i < DIM_; i += blockDim.x) dst[i] = src[i];
}

// ---------------------------------------------------------------------------
// RMSNorm: out_bf16 = x / max(||x||,1e-12) * sqrt(DIM) * gamma
// ---------------------------------------------------------------------------
__global__ __launch_bounds__(256) void rmsnorm_kernel(
    const float* __restrict__ x, const float* __restrict__ gamma,
    unsigned short* __restrict__ out) {
  int row = blockIdx.x;
  const float* xr = x + (size_t)row * DIM_;
  float ss = 0.f;
  for (int i = threadIdx.x; i < DIM_; i += 256) { float v = xr[i]; ss += v * v; }
  __shared__ float red[256];
  red[threadIdx.x] = ss;
  __syncthreads();
  for (int s = 128; s > 0; s >>= 1) {
    if (threadIdx.x < s) red[threadIdx.x] += red[threadIdx.x + s];
    __syncthreads();
  }
  float scale = 32.0f / fmaxf(sqrtf(red[0]), 1e-12f);  // sqrt(1024)=32
  unsigned short* o = out + (size_t)row * DIM_;
  for (int i = threadIdx.x; i < DIM_; i += 256)
    o[i] = f32_to_bf16(xr[i] * scale * gamma[i]);
}

// ---------------------------------------------------------------------------
// Rotary + split QKV -> head-major bf16 Q (pre-scaled by dh^-0.5), K, V
// ---------------------------------------------------------------------------
__global__ __launch_bounds__(512) void rotary_kernel(
    const float* __restrict__ qkv,
    unsigned short* __restrict__ Qb, unsigned short* __restrict__ Kb,
    unsigned short* __restrict__ Vb) {
  int n = blockIdx.x;
  int h = threadIdx.x >> 5;
  int i = threadIdx.x & 31;  // pair (i, i+32)
  float inv = __expf(-(2.0f * (float)i / (float)DH_) * 9.210340371976184f);
  float fr = (float)n * inv;
  float sn, cs;
  __sincosf(fr, &sn, &cs);
  const float* base = qkv + (size_t)n * (3 * DIM_);
  size_t ho = ((size_t)h * NSEQ + n) * DH_;
  {
    float t1 = base[h * DH_ + i];
    float t2 = base[h * DH_ + i + 32];
    Qb[ho + i]      = f32_to_bf16(0.125f * (t1 * cs - t2 * sn));
    Qb[ho + i + 32] = f32_to_bf16(0.125f * (t2 * cs + t1 * sn));
  }
  {
    float t1 = base[DIM_ + h * DH_ + i];
    float t2 = base[DIM_ + h * DH_ + i + 32];
    Kb[ho + i]      = f32_to_bf16(t1 * cs - t2 * sn);
    Kb[ho + i + 32] = f32_to_bf16(t2 * cs + t1 * sn);
  }
  {
    Vb[ho + i]      = f32_to_bf16(base[2 * DIM_ + h * DH_ + i]);
    Vb[ho + i + 32] = f32_to_bf16(base[2 * DIM_ + h * DH_ + i + 32]);
  }
}

// ---------------------------------------------------------------------------
// Tiled bf16 WMMA GEMM: C[M,N] (f32) = A[M,K] (bf16) * Bt[N,K] (bf16, pre-T)
// Block tile 128x128, 8 waves (wave tile 32x64 = 2x4 WMMA), BK=32, LDS
// double-buffered with async memory->LDS prefetch of tile k+1.
// Requires M%128==0, N%128==0, K%32==0 (true for all shapes here).
// ---------------------------------------------------------------------------
#define GBM 128
#define GBN 128
#define GBK 32

__device__ __forceinline__ void gemm_stage_tiles(
    const unsigned short* __restrict__ A, const unsigned short* __restrict__ Bt,
    unsigned short* lA, unsigned short* lB,
    int tid, int bm, int bn, int K, int kt) {
  int r = tid >> 1, c = (tid & 1) * 16;          // 128 rows x 32 K, 16 elems/thr
  const unsigned short* ga = A  + (size_t)(bm + r) * K + kt + c;
  const unsigned short* gb = Bt + (size_t)(bn + r) * K + kt + c;
#ifdef USE_ASYNC_LDS
  async_copy16(ga,     &lA[r * GBK + c]);
  async_copy16(ga + 8, &lA[r * GBK + c + 8]);
  async_copy16(gb,     &lB[r * GBK + c]);
  async_copy16(gb + 8, &lB[r * GBK + c + 8]);
#else
  uint4 a0 = *(const uint4*)ga, a1 = *(const uint4*)(ga + 8);
  uint4 b0 = *(const uint4*)gb, b1 = *(const uint4*)(gb + 8);
  *(uint4*)&lA[r * GBK + c]     = a0;
  *(uint4*)&lA[r * GBK + c + 8] = a1;
  *(uint4*)&lB[r * GBK + c]     = b0;
  *(uint4*)&lB[r * GBK + c + 8] = b1;
#endif
}

__global__ __launch_bounds__(256) void gemm_bf16_kernel(
    const unsigned short* __restrict__ A, const unsigned short* __restrict__ Bt,
    float* __restrict__ C, unsigned short* __restrict__ Cbf,
    const float* __restrict__ bias, const float* __restrict__ residual,
    int M, int N, int K, int gelu) {
  __shared__ __align__(16) unsigned short lA[2][GBM * GBK];  // [r][k]
  __shared__ __align__(16) unsigned short lB[2][GBN * GBK];  // [n][k]

  int tid  = threadIdx.x;
  int lane = tid & 31, wid = tid >> 5;
  int half = lane >> 4, l16 = lane & 15;
  int wm = (wid >> 1) * 32, wn = (wid & 1) * 64;
  int bm = blockIdx.y * GBM, bn = blockIdx.x * GBN;

  v8f acc[2][4];
#pragma unroll
  for (int mi = 0; mi < 2; ++mi)
#pragma unroll
    for (int ni = 0; ni < 4; ++ni) acc[mi][ni] = vzero8();

  int nk = K / GBK;
  gemm_stage_tiles(A, Bt, lA[0], lB[0], tid, bm, bn, K, 0);

  for (int ki = 0; ki < nk; ++ki) {
    int buf = ki & 1;
    wait_async0();          // this wave's async writes into lA/lB[buf] done
    __syncthreads();        // everyone's writes visible; prev reads finished
    if (ki + 1 < nk)
      gemm_stage_tiles(A, Bt, lA[buf ^ 1], lB[buf ^ 1], tid, bm, bn, K,
                       (ki + 1) * GBK);

    Frag af[2], bfr[4];
#pragma unroll
    for (int mi = 0; mi < 2; ++mi) {
      int row = wm + mi * 16 + l16;
      // A layout: lanes 0-15 hold K=0..15, lanes 16-31 hold K=16..31
      af[mi].u[0] = *(const uint4*)&lA[buf][row * GBK + half * 16];
      af[mi].u[1] = *(const uint4*)&lA[buf][row * GBK + half * 16 + 8];
    }
#pragma unroll
    for (int ni = 0; ni < 4; ++ni) {
      int col = wn + ni * 16 + l16;
      // B layout: lanes 0-15 hold K=0..15, lanes 16-31 hold K=16..31
      bfr[ni].u[0] = *(const uint4*)&lB[buf][col * GBK + half * 16];
      bfr[ni].u[1] = *(const uint4*)&lB[buf][col * GBK + half * 16 + 8];
    }
#pragma unroll
    for (int mi = 0; mi < 2; ++mi)
#pragma unroll
      for (int ni = 0; ni < 4; ++ni)
        acc[mi][ni] = wmma_bf16(af[mi].v, bfr[ni].v, acc[mi][ni]);
  }

  // Epilogue: C layout row = v + 8*half, col = l16 within each 16x16 tile.
#pragma unroll
  for (int mi = 0; mi < 2; ++mi)
#pragma unroll
    for (int ni = 0; ni < 4; ++ni)
#pragma unroll
      for (int v = 0; v < 8; ++v) {
        int row = bm + wm + mi * 16 + v + half * 8;
        int col = bn + wn + ni * 16 + l16;
        float val = acc[mi][ni][v];
        if (bias)     val += bias[col];
        if (gelu)     val  = 0.5f * val * (1.0f + erff(val * 0.70710678118f));
        if (residual) val += residual[(size_t)row * N + col];
        C[(size_t)row * N + col] = val;
        if (Cbf) Cbf[(size_t)row * N + col] = f32_to_bf16(val);
      }
}

// ---------------------------------------------------------------------------
// Flash-style causal attention.  Grid: (HEADS, N/128).  8 waves, each owns 16
// query rows; K/V tiles (32 keys) shared through LDS.  S and P*V on WMMA.
// ---------------------------------------------------------------------------
__global__ __launch_bounds__(256) void attn_kernel(
    const unsigned short* __restrict__ Qb, const unsigned short* __restrict__ Kb,
    const unsigned short* __restrict__ Vb, unsigned short* __restrict__ O) {
  __shared__ __align__(16) unsigned short lK[32 * DH_];   // [key][dh]
  __shared__ __align__(16) unsigned short lVt[DH_ * 32];  // [dh][key]
  __shared__ __align__(16) unsigned short lP[8][16 * 32]; // per-wave P tile

  int h = blockIdx.x, qb = blockIdx.y;
  int tid = threadIdx.x;
  int lane = tid & 31, wid = tid >> 5;
  int half = lane >> 4, l16 = lane & 15;
  int qr0 = qb * 128 + wid * 16;

  const unsigned short* Qh = Qb + (size_t)h * NSEQ * DH_;
  const unsigned short* Kh = Kb + (size_t)h * NSEQ * DH_;
  const unsigned short* Vh = Vb + (size_t)h * NSEQ * DH_;

  Frag aq[2];  // Q fragments reused over all key blocks, dh windows 0/32
#pragma unroll
  for (int w = 0; w < 2; ++w) {
    const unsigned short* qr = Qh + (size_t)(qr0 + l16) * DH_ + w * 32;
    aq[w].u[0] = *(const uint4*)(qr + half * 16);
    aq[w].u[1] = *(const uint4*)(qr + half * 16 + 8);
  }

  float m_run[8], l_run[8];
  v8f oacc[4];
#pragma unroll
  for (int v = 0; v < 8; ++v) { m_run[v] = -1e30f; l_run[v] = 0.f; }
#pragma unroll
  for (int ni = 0; ni < 4; ++ni) oacc[ni] = vzero8();

  int nkb = qb * 4 + 4;
  for (int kb = 0; kb < nkb; ++kb) {
    int key0 = kb * 32;
    __syncthreads();
    {
      int r = tid >> 3, c = (tid & 7) * 8;   // 32 keys x 64 dh
#ifdef USE_ASYNC_LDS
      async_copy16(Kh + (size_t)(key0 + r) * DH_ + c, &lK[r * DH_ + c]);
#else
      uint4 kq = *(const uint4*)(Kh + (size_t)(key0 + r) * DH_ + c);
      *(uint4*)&lK[r * DH_ + c] = kq;
#endif
      uint4 vq = *(const uint4*)(Vh + (size_t)(key0 + r) * DH_ + c);
      unsigned short e[8];
      e[0] = vq.x & 0xffff; e[1] = vq.x >> 16;
      e[2] = vq.y & 0xffff; e[3] = vq.y >> 16;
      e[4] = vq.z & 0xffff; e[5] = vq.z >> 16;
      e[6] = vq.w & 0xffff; e[7] = vq.w >> 16;
#pragma unroll
      for (int j = 0; j < 8; ++j) lVt[(c + j) * 32 + r] = e[j];
      wait_async0();
    }
    __syncthreads();

    // S = Q K^T for 32 keys: two 16x16 C tiles, 2 WMMAs each over dh
    v8f s0 = vzero8(), s1 = vzero8();
#pragma unroll
    for (int w = 0; w < 2; ++w) {
      Frag bk0, bk1;
      bk0.u[0] = *(const uint4*)&lK[(l16)      * DH_ + w * 32 + half * 16];
      bk0.u[1] = *(const uint4*)&lK[(l16)      * DH_ + w * 32 + half * 16 + 8];
      bk1.u[0] = *(const uint4*)&lK[(l16 + 16) * DH_ + w * 32 + half * 16];
      bk1.u[1] = *(const uint4*)&lK[(l16 + 16) * DH_ + w * 32 + half * 16 + 8];
      s0 = wmma_bf16(aq[w].v, bk0.v, s0);
      s1 = wmma_bf16(aq[w].v, bk1.v, s1);
    }

    // Causal mask + online softmax (row values across 16 lanes per vgpr)
#pragma unroll
    for (int v = 0; v < 8; ++v) {
      int row  = qr0 + v + half * 8;
      int col0 = key0 + l16;
      int col1 = col0 + 16;
      float a = (col0 > row) ? -1e30f : s0[v];
      float b = (col1 > row) ? -1e30f : s1[v];
      float mloc = fmaxf(a, b);
#pragma unroll
      for (int off = 1; off < 16; off <<= 1)
        mloc = fmaxf(mloc, __shfl_xor(mloc, off, 16));
      float mnew = fmaxf(m_run[v], mloc);
      float rescale = __expf(m_run[v] - mnew);
      float p0 = __expf(a - mnew);
      float p1 = __expf(b - mnew);
      float lsum = p0 + p1;
#pragma unroll
      for (int off = 1; off < 16; off <<= 1)
        lsum += __shfl_xor(lsum, off, 16);
      l_run[v] = l_run[v] * rescale + lsum;
      m_run[v] = mnew;
#pragma unroll
      for (int ni = 0; ni < 4; ++ni) oacc[ni][v] *= rescale;
      lP[wid][(v + half * 8) * 32 + l16]      = f32_to_bf16(p0);
      lP[wid][(v + half * 8) * 32 + l16 + 16] = f32_to_bf16(p1);
    }

    // P (16x32, A layout) times V (32 x 64, B from transposed lVt)
    Frag pa;
    pa.u[0] = *(const uint4*)&lP[wid][l16 * 32 + half * 16];
    pa.u[1] = *(const uint4*)&lP[wid][l16 * 32 + half * 16 + 8];
#pragma unroll
    for (int ni = 0; ni < 4; ++ni) {
      Frag bv;
      int dcol = ni * 16 + l16;
      bv.u[0] = *(const uint4*)&lVt[dcol * 32 + half * 16];
      bv.u[1] = *(const uint4*)&lVt[dcol * 32 + half * 16 + 8];
      oacc[ni] = wmma_bf16(pa.v, bv.v, oacc[ni]);
    }
  }

#pragma unroll
  for (int v = 0; v < 8; ++v) {
    float invl = 1.0f / l_run[v];
    int row = qr0 + v + half * 8;
#pragma unroll
    for (int ni = 0; ni < 4; ++ni) {
      int col = h * DH_ + ni * 16 + l16;
      O[(size_t)row * DIM_ + col] = f32_to_bf16(oacc[ni][v] * invl);
    }
  }
}

// ---------------------------------------------------------------------------
// Host orchestration
// ---------------------------------------------------------------------------
static inline size_t alignup(size_t x) { return (x + 255) & ~(size_t)255; }

extern "C" void kernel_launch(void* const* d_in, const int* in_sizes, int n_in,
                              void* d_out, int out_size, void* d_ws, size_t ws_size,
                              hipStream_t stream) {
  (void)in_sizes; (void)n_in; (void)out_size; (void)ws_size;
  const int*   tokens   = (const int*)  d_in[0];
  const float* emb      = (const float*)d_in[1];
  const float* g_attn   = (const float*)d_in[2];
  const float* w_qkv    = (const float*)d_in[3];
  const float* w_out    = (const float*)d_in[4];
  const float* g_ff     = (const float*)d_in[5];
  const float* w_ff1    = (const float*)d_in[6];
  const float* b_ff1    = (const float*)d_in[7];
  const float* w_ff2    = (const float*)d_in[8];
  const float* b_ff2    = (const float*)d_in[9];
  const float* g_final  = (const float*)d_in[10];
  const float* w_logits = (const float*)d_in[11];
  float* out = (float*)d_out;

  char* p = (char*)d_ws;
  size_t off = 0;
  auto take = [&](size_t bytes) { char* q = p + off; off += alignup(bytes); return q; };
  float*          xf   = (float*)         take((size_t)NSEQ * DIM_ * 4);
  unsigned short* xn   = (unsigned short*)take((size_t)NSEQ * DIM_ * 2);
  float*          qkv  = (float*)         take((size_t)NSEQ * 3 * DIM_ * 4);
  unsigned short* Qb   = (unsigned short*)take((size_t)HEADS_ * NSEQ * DH_ * 2);
  unsigned short* Kb   = (unsigned short*)take((size_t)HEADS_ * NSEQ * DH_ * 2);
  unsigned short* Vb   = (unsigned short*)take((size_t)HEADS_ * NSEQ * DH_ * 2);
  unsigned short* ao   = (unsigned short*)take((size_t)NSEQ * DIM_ * 2);
  float*          ff   = (float*)         take((size_t)NSEQ * FF_ * 4);
  unsigned short* ffb  = (unsigned short*)take((size_t)NSEQ * FF_ * 2);
  unsigned short* wqkvb = (unsigned short*)take((size_t)DEPTH_ * DIM_ * 3 * DIM_ * 2);
  unsigned short* woutb = (unsigned short*)take((size_t)DEPTH_ * DIM_ * DIM_ * 2);
  unsigned short* wff1b = (unsigned short*)take((size_t)DEPTH_ * DIM_ * FF_ * 2);
  unsigned short* wff2b = (unsigned short*)take((size_t)DEPTH_ * FF_ * DIM_ * 2);
  unsigned short* wlogb = (unsigned short*)take((size_t)DIM_ * VOCAB_ * 2);

  // weights: fp32 [K][N] -> bf16 [N][K], once per call
  auto cvtT = [&](const float* src, unsigned short* dst, int K, int N) {
    cvt_transpose_kernel<<<dim3(N / 32, K / 32), dim3(32, 8), 0, stream>>>(
        src, dst, K, N);
  };
  for (int l = 0; l < DEPTH_; ++l) {
    cvtT(w_qkv + (size_t)l * DIM_ * 3 * DIM_, wqkvb + (size_t)l * DIM_ * 3 * DIM_,
         DIM_, 3 * DIM_);
    cvtT(w_out + (size_t)l * DIM_ * DIM_, woutb + (size_t)l * DIM_ * DIM_,
         DIM_, DIM_);
    cvtT(w_ff1 + (size_t)l * DIM_ * FF_, wff1b + (size_t)l * DIM_ * FF_,
         DIM_, FF_);
    cvtT(w_ff2 + (size_t)l * FF_ * DIM_, wff2b + (size_t)l * FF_ * DIM_,
         FF_, DIM_);
  }
  cvtT(w_logits, wlogb, DIM_, VOCAB_);

  embed_kernel<<<NSEQ, 256, 0, stream>>>(tokens, emb, xf);

  auto gemm = [&](const unsigned short* A, const unsigned short* Bt, float* C,
                  unsigned short* Cbf, const float* bias, const float* res,
                  int M, int N, int K, int gelu) {
    dim3 grid(N / GBN, M / GBM);
    gemm_bf16_kernel<<<grid, 256, 0, stream>>>(A, Bt, C, Cbf, bias, res, M, N, K, gelu);
  };

  for (int l = 0; l < DEPTH_; ++l) {
    rmsnorm_kernel<<<NSEQ, 256, 0, stream>>>(xf, g_attn + (size_t)l * DIM_, xn);
    gemm(xn, wqkvb + (size_t)l * DIM_ * 3 * DIM_, qkv, nullptr, nullptr, nullptr,
         NSEQ, 3 * DIM_, DIM_, 0);
    rotary_kernel<<<NSEQ, 512, 0, stream>>>(qkv, Qb, Kb, Vb);
    attn_kernel<<<dim3(HEADS_, NSEQ / 128), 256, 0, stream>>>(Qb, Kb, Vb, ao);
    gemm(ao, woutb + (size_t)l * DIM_ * DIM_, xf, nullptr, nullptr, xf,
         NSEQ, DIM_, DIM_, 0);
    rmsnorm_kernel<<<NSEQ, 256, 0, stream>>>(xf, g_ff + (size_t)l * DIM_, xn);
    gemm(xn, wff1b + (size_t)l * DIM_ * FF_, ff, ffb,
         b_ff1 + (size_t)l * FF_, nullptr, NSEQ, FF_, DIM_, 1);
    gemm(ffb, wff2b + (size_t)l * FF_ * DIM_, xf, nullptr,
         b_ff2 + (size_t)l * DIM_, xf, NSEQ, DIM_, FF_, 0);
  }

  rmsnorm_kernel<<<NSEQ, 256, 0, stream>>>(xf, g_final, xn);
  gemm(xn, wlogb, out, nullptr, nullptr, nullptr, NSEQ, VOCAB_, DIM_, 0);
}
